// LDC_78219944395333
// MI455X (gfx1250) — matrix operand order repaired and checked by reading, not compiled, not run on hardware
//
#include <hip/hip_runtime.h>
#include <hip/hip_bf16.h>

// ---------------------------------------------------------------------------
// Problem constants
#define B_   16
#define CI_  64
#define CO_  64
#define H_   128
#define W_   128
#define KT_  576            // CI*3*3, GEMM K, ordered k' = (kh*3+kw)*64 + ci
#define NPIX (H_*W_)

typedef __attribute__((ext_vector_type(16))) __bf16 v16bf;
typedef __attribute__((ext_vector_type(8)))  __bf16 v8bf;
typedef __attribute__((ext_vector_type(8)))  float  v8f;

// ---------------------------------------------------------------------------
// Stage 1: per (b, ci) mean / unbiased std over 128x128 spatial
__global__ __launch_bounds__(256) void stats_kernel(const float* __restrict__ x,
                                                    float* __restrict__ stats) {
    int b = blockIdx.x >> 6;
    int c = blockIdx.x & 63;
    const float* p = x + (size_t)(b * CI_ + c) * NPIX;
    float s = 0.f, s2 = 0.f;
    for (int i = threadIdx.x; i < NPIX; i += 256) {
        float v = p[i];
        s += v; s2 += v * v;
    }
    __shared__ float red0[256];
    __shared__ float red1[256];
    red0[threadIdx.x] = s; red1[threadIdx.x] = s2;
    __syncthreads();
    for (int off = 128; off > 0; off >>= 1) {
        if (threadIdx.x < off) {
            red0[threadIdx.x] += red0[threadIdx.x + off];
            red1[threadIdx.x] += red1[threadIdx.x + off];
        }
        __syncthreads();
    }
    if (threadIdx.x == 0) {
        float sum = red0[0], sumsq = red1[0];
        const float n = (float)NPIX;
        float mean = sum / n;
        float var  = (sumsq - sum * sum / n) / (n - 1.f);   // ddof = 1
        stats[b * 128 + c]      = mean;
        stats[b * 128 + 64 + c] = sqrtf(fmaxf(var, 0.f));
    }
}

// ---------------------------------------------------------------------------
// Stage 2a: h = relu(stats @ w1 + b1)   [16,128]@[128,32]
__global__ void mlp1_kernel(const float* __restrict__ stats,
                            const float* __restrict__ w1,
                            const float* __restrict__ b1,
                            float* __restrict__ h) {
    int b = blockIdx.x;           // 16 blocks
    int s = threadIdx.x;          // 32 threads
    float acc = b1[s];
    for (int k = 0; k < 128; ++k)
        acc += stats[b * 128 + k] * w1[k * 32 + s];
    h[b * 32 + s] = fmaxf(acc, 0.f);
}

// ---------------------------------------------------------------------------
// Stage 2b: kernels = h @ w2 + b2, written as bf16 in WMMA-friendly K order:
//   wbf[b][co][k'] with k' = (kh*3+kw)*64 + ci
__global__ __launch_bounds__(256) void mlp2_kernel(const float* __restrict__ h,
                                                   const float* __restrict__ w2,
                                                   const float* __restrict__ b2,
                                                   __bf16* __restrict__ wbf) {
    int t = blockIdx.x * 256 + threadIdx.x;  // 2304*256 == 16*36864 exactly
    int b = t / 36864;
    int j = t % 36864;                        // j = (co*64+ci)*9 + (kh*3+kw)
    float acc = b2[j];
    const float* hb = h + b * 32;
#pragma unroll
    for (int s = 0; s < 32; ++s)
        acc += hb[s] * w2[s * 36864 + j];
    int co = j / 576;
    int r  = j % 576;
    int ci = r / 9;
    int q  = r % 9;
    wbf[((size_t)b * CO_ + co) * KT_ + q * 64 + ci] = (__bf16)acc;
}

// ---------------------------------------------------------------------------
// Stage 3: implicit-GEMM conv via v_wmma_f32_16x16x32_bf16.
// One workgroup (8 waves) per (b, ph0): computes out[b, 0:64, ph0:ph0+2, 0:128].
// LDS slab: x rows ph0-1..ph0+2, layout [r(4)][col(130, zero-padded)][ci(64)] bf16
// (channel-fastest so every WMMA B fragment is two contiguous ds_load_b128).
__global__ __launch_bounds__(256) void conv_wmma_kernel(const float* __restrict__ x,
                                                        const __bf16* __restrict__ wbf,
                                                        float* __restrict__ out) {
    __shared__ __align__(16) __bf16 lds_x[4 * 130 * 64];   // 66,560 B (CDNA5: 320KB/WG)

    int b   = blockIdx.x >> 6;
    int ph0 = (blockIdx.x & 63) * 2;
    int tid = threadIdx.x;

    // zero the two padding columns (col index 0 and 129) for all 4 slab rows
    for (int i = tid; i < 4 * 64 * 2; i += 256) {
        int r = i >> 7, rem = i & 127;
        int ci = rem >> 1;
        int c  = (rem & 1) ? 129 : 0;
        lds_x[(r * 130 + c) * 64 + ci] = (__bf16)0.0f;
    }
    // interior fill, float4-vectorized: lds_x[(r*130 + c+1)*64 + ci] = x[b,ci,ph0+r-1,c]
    for (int i = tid; i < 4 * 64 * 32; i += 256) {   // 8192 quads
        int r   = i >> 11;          // /(64*32)
        int rem = i & 2047;
        int ci  = rem >> 5;
        int c4  = (rem & 31) << 2;
        int row = ph0 + r - 1;
        float4 v = make_float4(0.f, 0.f, 0.f, 0.f);
        if (row >= 0 && row < H_)
            v = *(const float4*)(x + (((size_t)b * CI_ + ci) * H_ + row) * W_ + c4);
        int base = (r * 130 + c4 + 1) * 64 + ci;
        lds_x[base      ] = (__bf16)v.x;
        lds_x[base +  64] = (__bf16)v.y;
        lds_x[base + 128] = (__bf16)v.z;
        lds_x[base + 192] = (__bf16)v.w;
    }
    __syncthreads();

    int wave  = tid >> 5;
    int lane  = tid & 31;
    int lhalf = lane >> 4;          // 0 | 1
    int l16   = lane & 15;
    int m0 = (wave & 3) * 16;       // CO sub-block
    int n0 = (wave >> 2) * 64;      // pixel sub-block

    v8f acc[2][4] = {{v8f{}, v8f{}, v8f{}, v8f{}},
                     {v8f{}, v8f{}, v8f{}, v8f{}}};
    const __bf16* wrow = wbf + (size_t)b * CO_ * KT_;

    for (int ks = 0; ks < 18; ++ks) {
        int q   = ks >> 1;          // (kh*3+kw), constant within a K-step
        int ci0 = (ks & 1) * 32;
        int kh  = q / 3, kw = q % 3;

        // ---- A fragment (16x32 bf16): ISA layout
        //   lanes 0-15 : K = 0-7 (v0-3), 16-23 (v4-7)
        //   lanes 16-31: K = 8-15,        24-31
        int m = m0 + l16;
        const __bf16* ap = wrow + (size_t)m * KT_ + ks * 32 + (lhalf ? 8 : 0);
        v8bf a_lo = *(const v8bf*)(ap);        // k_local +0..7
        v8bf a_hi = *(const v8bf*)(ap + 16);   // k_local +16..23
        v16bf afrag = __builtin_shufflevector(a_lo, a_hi,
            0,1,2,3,4,5,6,7, 8,9,10,11,12,13,14,15);

        // ---- B fragments (32x16 bf16): lanes 0-15 hold K 0-7,16-23, lanes 16-31 K 8-15,24-31
        // consecutive K == consecutive ci at fixed (kh,kw,pixel) -> contiguous in LDS
        int cistart = ci0 + (lhalf ? 8 : 0);
#pragma unroll
        for (int rr = 0; rr < 2; ++rr) {        // two output rows share this A frag
            int rowbase = ((kh + rr) * 130 + kw) * 64 + cistart;
#pragma unroll
            for (int t = 0; t < 4; ++t) {
                int n = n0 + t * 16 + l16;      // pixel column
                const __bf16* bp = &lds_x[rowbase + n * 64];
                v8bf b_lo = *(const v8bf*)(bp);
                v8bf b_hi = *(const v8bf*)(bp + 16);
                v16bf bfrag = __builtin_shufflevector(b_lo, b_hi,
                    0,1,2,3,4,5,6,7, 8,9,10,11,12,13,14,15);
                acc[rr][t] = __builtin_amdgcn_wmma_f32_16x16x32_bf16(
                    false, afrag, false, bfrag, (short)0, acc[rr][t], false, false);
            }
        }
    }

    // ---- store: C/D layout VGPR i -> M = i (lanes 0-15) / i+8 (lanes 16-31)
#pragma unroll
    for (int rr = 0; rr < 2; ++rr) {
        int ph = ph0 + rr;
#pragma unroll
        for (int t = 0; t < 4; ++t) {
            int n = n0 + t * 16 + l16;
#pragma unroll
            for (int i = 0; i < 8; ++i) {
                int co = m0 + i + lhalf * 8;
                out[(((size_t)b * CO_ + co) * H_ + ph) * W_ + n] = acc[rr][t][i];
            }
        }
    }
}

// ---------------------------------------------------------------------------
extern "C" void kernel_launch(void* const* d_in, const int* in_sizes, int n_in,
                              void* d_out, int out_size, void* d_ws, size_t ws_size,
                              hipStream_t stream) {
    const float* x  = (const float*)d_in[0];
    const float* w1 = (const float*)d_in[1];
    const float* b1 = (const float*)d_in[2];
    const float* w2 = (const float*)d_in[3];
    const float* b2 = (const float*)d_in[4];
    float* out = (float*)d_out;

    char* ws = (char*)d_ws;
    float*  stats = (float*)ws;                    //  8 KB : [16,128]
    float*  h     = (float*)(ws + 8192);           //  2 KB : [16,32]
    __bf16* wbf   = (__bf16*)(ws + 16384);         // ~1.13 MB : [16,64,576] bf16

    stats_kernel    <<<B_ * CI_, 256, 0, stream>>>(x, stats);
    mlp1_kernel     <<<B_, 32,        0, stream>>>(stats, w1, b1, h);
    mlp2_kernel     <<<2304, 256,     0, stream>>>(h, w2, b2, wbf);
    conv_wmma_kernel<<<B_ * (H_ / 2), 256, 0, stream>>>(x, wbf, out);
}
